// Decoder_70360154243533
// MI455X (gfx1250) — matrix operand order repaired and checked by reading, not compiled, not run on hardware
//
#include <hip/hip_runtime.h>
#include <stdint.h>

// ---------------------------------------------------------------------------
// LSTM decoder: B=32, T=128, D=512, U=1024 (4U=4096), V=32000
// bf16 WMMA (v_wmma_f32_16x16x32_bf16) for all GEMMs, fp32 accumulate.
// Big GEMMs use LDS-staged, double-buffered B tiles shared by 8 waves
// (async load-to-LDS when the toolchain exposes the CDNA5 async path).
// ---------------------------------------------------------------------------

#define LB   32
#define LT   128
#define LD   512
#define LU   1024
#define LG4  4096
#define LV   32000
#define LOGITS_ELEMS (LB * LT * LV)   // 131,072,000

#if defined(__gfx1250__) && __has_builtin(__builtin_amdgcn_s_wait_asynccnt)
#define USE_ASYNC_LDS 1
#else
#define USE_ASYNC_LDS 0
#endif

typedef __attribute__((ext_vector_type(16))) __bf16 v16bf;
typedef __attribute__((ext_vector_type(8)))  float  v8f;

struct FragBits { uint4 lo, hi; };   // 32 bytes == sizeof(v16bf)

__device__ __forceinline__ unsigned short f2bf(float f) {
  unsigned int u = __float_as_uint(f);
  u += 0x7FFFu + ((u >> 16) & 1u);   // round-to-nearest-even
  return (unsigned short)(u >> 16);
}

// A fragment: 16x32 bf16, row-major source A[m][k] (lda elems).
// lanes 0-15 -> M=lane; VGPR0..3 = K(h*8 + 0..7), VGPR4..7 = K(16 + h*8 + 0..7)
__device__ __forceinline__ v16bf load_frag_a(const unsigned short* __restrict__ A,
                                             int lda, int lane) {
  const int m = lane & 15, h = lane >> 4;
  const unsigned short* p = A + (size_t)m * lda + h * 8;
  FragBits fb;
  fb.lo = *(const uint4*)(p);
  fb.hi = *(const uint4*)(p + 16);
  return __builtin_bit_cast(v16bf, fb);
}

// B fragment from [N][K]-layout bf16 source with row stride ldb elems.
// lane -> N=lane&15; VGPR0..3 = K(h*8 + 0..7), VGPR4..7 = K(16 + h*8 + 0..7)
__device__ __forceinline__ v16bf load_frag_b(const unsigned short* __restrict__ Bt,
                                             int ldb, int lane) {
  const int n = lane & 15, h = lane >> 4;
  const unsigned short* p = Bt + (size_t)n * ldb + h * 8;
  FragBits fb;
  fb.lo = *(const uint4*)(p);
  fb.hi = *(const uint4*)(p + 16);
  return __builtin_bit_cast(v16bf, fb);
}

#if USE_ASYNC_LDS
// CDNA5 async copy: global -> LDS, tracked by ASYNCcnt.
__device__ __forceinline__ void async_copy_b128(unsigned int lds_off,
                                                const void* gptr) {
  asm volatile("global_load_async_to_lds_b128 %0, %1, off"
               :: "v"(lds_off), "v"((unsigned long long)(uintptr_t)gptr)
               : "memory");
}
#endif

// ---------------------------------------------------------------------------
// fp32 -> bf16 elementwise convert
// ---------------------------------------------------------------------------
__global__ void k_cvt_bf16(const float* __restrict__ in,
                           unsigned short* __restrict__ out, int n) {
  int i = blockIdx.x * blockDim.x + threadIdx.x;
  if (i < n) out[i] = f2bf(in[i]);
}

// fp32 [K][N] -> bf16 [N][K] tiled transpose (both sides coalesced)
__global__ void __launch_bounds__(256) k_transpose_cvt(const float* __restrict__ in,
                                                       unsigned short* __restrict__ out,
                                                       int K, int N) {
  __shared__ float tile[32][33];
  const int nb = blockIdx.x * 32, kb = blockIdx.y * 32;
  const int tx = threadIdx.x, ty = threadIdx.y;   // 32 x 8
#pragma unroll
  for (int i = 0; i < 32; i += 8)
    tile[ty + i][tx] = in[(size_t)(kb + ty + i) * N + nb + tx];
  __syncthreads();
#pragma unroll
  for (int i = 0; i < 32; i += 8)
    out[(size_t)(nb + ty + i) * K + kb + tx] = f2bf(tile[tx][ty + i]);
}

// ---------------------------------------------------------------------------
// Main WMMA GEMM with LDS-staged B: C[M][N] = A@Bt^T + biasN.
// Block = 256 threads (8 waves), block tile 128(M) x 64(N).
// B tile 64x32 bf16 staged in LDS (double-buffered), shared by all 8 waves.
// ---------------------------------------------------------------------------
#define KC   32          // K chunk per stage
#define BM   128
#define BN   64
#define BSTR 40          // LDS row stride (elements): 80B, spreads banks

__global__ void __launch_bounds__(256) k_gemm_bf16_lds(
    const unsigned short* __restrict__ A, const unsigned short* __restrict__ Bt,
    const float* __restrict__ biasN, float* __restrict__ C,
    int M, int N, int K) {
  __shared__ unsigned short btile[2][BN * BSTR];   // 2 x 5 KB

  const int tid  = threadIdx.x;
  const int lane = tid & 31;
  const int wv   = tid >> 5;                       // 0..7 -> M sub-tile
  const int nblk = N / BN;
  const int bidm = blockIdx.x / nblk;
  const int bidn = blockIdx.x - bidm * nblk;
  const int bm   = bidm * BM, bn = bidn * BN;

  // Staging map: thread -> (row 0..63, 8-element chunk 0..3) of the 64x32 tile
  const int srow = tid >> 2;
  const int scol = (tid & 3) * 8;
  const unsigned short* gsrc = Bt + (size_t)(bn + srow) * K + scol;
  unsigned short* l0 = &btile[0][srow * BSTR + scol];
  unsigned short* l1 = &btile[1][srow * BSTR + scol];

  // prologue: stage kc = 0 into buffer 0
#if USE_ASYNC_LDS
  async_copy_b128((unsigned int)(uintptr_t)l0, gsrc);
#else
  *(uint4*)l0 = *(const uint4*)gsrc;
#endif

  v8f zero = {};
  v8f acc[4] = {zero, zero, zero, zero};
  const unsigned short* Arow = A + (size_t)(bm + wv * 16) * K;

  for (int kc = 0; kc < K; kc += KC) {
    const int cur = (kc >> 5) & 1;
#if USE_ASYNC_LDS
    __builtin_amdgcn_s_wait_asynccnt(0);           // my stage of buf[cur] done
    __syncthreads();                               // everyone's stage done
    if (kc + KC < K)                               // overlap next stage w/ math
      async_copy_b128((unsigned int)(uintptr_t)(cur ? l0 : l1), gsrc + kc + KC);
#else
    uint4 stg;
    const bool more = kc + KC < K;
    if (more) stg = *(const uint4*)(gsrc + kc + KC);   // issue early
    __syncthreads();                               // buf[cur] stores visible
#endif
    v16bf a = load_frag_a(Arow + kc, K, lane);
    const unsigned short* tcur = btile[cur];
#pragma unroll
    for (int j = 0; j < 4; ++j) {
      v16bf b = load_frag_b(tcur + j * 16 * BSTR, BSTR, lane);  // ds_load_b128 x2
      acc[j] = __builtin_amdgcn_wmma_f32_16x16x32_bf16(
          false, a, false, b, (short)0, acc[j], false, false);
    }
#if !USE_ASYNC_LDS
    if (more) *(uint4*)(cur ? l0 : l1) = stg;      // ds_store_b128 into next buf
#endif
  }

  const int half = lane >> 4, nn = lane & 15;
#pragma unroll
  for (int j = 0; j < 4; ++j) {
    const int n = bn + j * 16 + nn;
    const float bv = biasN[n];
#pragma unroll
    for (int r = 0; r < 8; ++r) {
      const int m = bm + wv * 16 + half * 8 + r;   // C layout: VGPR r -> M=r (+8 hi)
      C[(size_t)m * N + n] = acc[j][r] + bv;
    }
  }
}

// ---------------------------------------------------------------------------
// Recurrent GEMM (one timestep): z[b][n] = xW[b*T+t][n] + (h*m_g) @ U_g
// M=32, N=4096, K=1024 -> 128 waves. U is L2-resident; direct loads.
// ---------------------------------------------------------------------------
__global__ void __launch_bounds__(256) k_rec_gemm(
    const unsigned short* __restrict__ hm, const unsigned short* __restrict__ Ut,
    const float* __restrict__ xW, float* __restrict__ z, int t) {
  const int lane = threadIdx.x & 31;
  const int wave = (blockIdx.x * 256 + threadIdx.x) >> 5;  // 0..127
  const int mt   = wave >> 6;        // 0..1
  const int nt   = wave & 63;        // 0..63
  const int gate = nt >> 4;          // 64-wide tile stays inside one gate

  v8f zero = {};
  v8f acc[4] = {zero, zero, zero, zero};
  const unsigned short* Abase = hm + (size_t)gate * LB * LU + (size_t)(mt * 16) * LU;
  const unsigned short* Bbase = Ut + (size_t)(nt * 64) * LU;

  for (int k0 = 0; k0 < LU; k0 += 32) {
    v16bf a = load_frag_a(Abase + k0, LU, lane);
#pragma unroll
    for (int j = 0; j < 4; ++j) {
      v16bf b = load_frag_b(Bbase + (size_t)(j * 16) * LU + k0, LU, lane);
      acc[j] = __builtin_amdgcn_wmma_f32_16x16x32_bf16(
          false, a, false, b, (short)0, acc[j], false, false);
    }
  }
  const int half = lane >> 4, nn = lane & 15;
#pragma unroll
  for (int j = 0; j < 4; ++j) {
#pragma unroll
    for (int r = 0; r < 8; ++r) {
      const int b = mt * 16 + half * 8 + r;
      const int n = nt * 64 + j * 16 + nn;
      z[(size_t)b * LG4 + n] = acc[j][r] + xW[((size_t)b * LT + t) * LG4 + n];
    }
  }
}

// ---------------------------------------------------------------------------
// Pointwise LSTM cell + emit bf16 h (hs) and pre-masked h (hm) for next step
// ---------------------------------------------------------------------------
__global__ void k_cell(const float* __restrict__ z, const float* __restrict__ masks,
                       float* __restrict__ c_ws, float* __restrict__ h_ws,
                       unsigned short* __restrict__ hm, unsigned short* __restrict__ hs,
                       int t) {
  const int idx = blockIdx.x * blockDim.x + threadIdx.x;   // 0..32767
  if (idx >= LB * LU) return;
  const int b = idx >> 10, u = idx & (LU - 1);
  const float* zb = z + (size_t)b * LG4;
  const float zi = zb[u], zf = zb[LU + u], zc = zb[2 * LU + u], zo = zb[3 * LU + u];
  const float ig = 1.0f / (1.0f + __expf(-zi));
  const float fg = 1.0f / (1.0f + __expf(-zf));
  const float og = 1.0f / (1.0f + __expf(-zo));
  const float cn = fg * c_ws[idx] + ig * tanhf(zc);
  const float hn = og * tanhf(cn);
  c_ws[idx] = cn;
  h_ws[idx] = hn;
  hs[((size_t)b * LT + t) * LU + u] = f2bf(hn);
#pragma unroll
  for (int g = 0; g < 4; ++g)
    hm[(size_t)g * LB * LU + idx] = f2bf(hn * masks[(size_t)g * LB * LU + idx]);
}

__global__ void k_init(const float* __restrict__ h0, const float* __restrict__ c0,
                       const float* __restrict__ masks,
                       unsigned short* __restrict__ hm,
                       float* __restrict__ c_ws, float* __restrict__ h_ws) {
  const int idx = blockIdx.x * blockDim.x + threadIdx.x;
  if (idx >= LB * LU) return;
  c_ws[idx] = c0[idx];
  h_ws[idx] = h0[idx];
#pragma unroll
  for (int g = 0; g < 4; ++g)
    hm[(size_t)g * LB * LU + idx] = f2bf(h0[idx] * masks[(size_t)g * LB * LU + idx]);
}

__global__ void k_finalize(const float* __restrict__ h_ws, const float* __restrict__ c_ws,
                           float* __restrict__ out_tail) {
  const int idx = blockIdx.x * blockDim.x + threadIdx.x;
  if (idx < LB * LU)           out_tail[idx] = h_ws[idx];
  else if (idx < 2 * LB * LU)  out_tail[idx] = c_ws[idx - LB * LU];
}

// ---------------------------------------------------------------------------
extern "C" void kernel_launch(void* const* d_in, const int* in_sizes, int n_in,
                              void* d_out, int out_size, void* d_ws, size_t ws_size,
                              hipStream_t stream) {
  const float* X      = (const float*)d_in[0];  // [B,T,D]
  const float* h0     = (const float*)d_in[1];  // [B,U]
  const float* c0     = (const float*)d_in[2];  // [B,U]
  const float* W      = (const float*)d_in[3];  // [D,4U]
  const float* Umat   = (const float*)d_in[4];  // [U,4U]
  const float* bvec   = (const float*)d_in[5];  // [4U]
  const float* Wd     = (const float*)d_in[6];  // [U,V]
  const float* bd     = (const float*)d_in[7];  // [V]
  const float* masks  = (const float*)d_in[8];  // [4,B,U]
  float* out = (float*)d_out;

  // ---- workspace layout (bytes, 256-aligned) ----
  char* ws = (char*)d_ws;
  size_t off = 0;
  auto alloc = [&](size_t bytes) -> char* {
    char* p = ws + off;
    off += (bytes + 255) & ~(size_t)255;
    return p;
  };
  unsigned short* Xb  = (unsigned short*)alloc((size_t)LB * LT * LD * 2);   //  4 MB
  unsigned short* Wt  = (unsigned short*)alloc((size_t)LG4 * LD * 2);       //  4 MB
  unsigned short* Ut  = (unsigned short*)alloc((size_t)LG4 * LU * 2);       //  8 MB
  unsigned short* Wdt = (unsigned short*)alloc((size_t)LV * LU * 2);        // 64 MB
  float*          xW  = (float*)alloc((size_t)LB * LT * LG4 * 4);           // 64 MB
  unsigned short* hs  = (unsigned short*)alloc((size_t)LB * LT * LU * 2);   //  8 MB
  unsigned short* hm  = (unsigned short*)alloc((size_t)4 * LB * LU * 2);
  float*          zbuf = (float*)alloc((size_t)LB * LG4 * 4);
  float*          h_ws = (float*)alloc((size_t)LB * LU * 4);
  float*          c_ws = (float*)alloc((size_t)LB * LU * 4);
  (void)ws_size; (void)in_sizes; (void)n_in; (void)out_size;

  // ---- phase 0: bf16 conversions / weight transposes ----
  {
    const int n = LB * LT * LD;
    k_cvt_bf16<<<(n + 255) / 256, 256, 0, stream>>>(X, Xb, n);
  }
  k_transpose_cvt<<<dim3(LG4 / 32, LD / 32), dim3(32, 8), 0, stream>>>(W, Wt, LD, LG4);
  k_transpose_cvt<<<dim3(LG4 / 32, LU / 32), dim3(32, 8), 0, stream>>>(Umat, Ut, LU, LG4);
  k_transpose_cvt<<<dim3(LV / 32, LU / 32), dim3(32, 8), 0, stream>>>(Wd, Wdt, LU, LV);

  // ---- phase 1: xW = X @ W + b   (M=4096, N=4096, K=512) ----
  k_gemm_bf16_lds<<<(LB * LT / BM) * (LG4 / BN), 256, 0, stream>>>(
      Xb, Wt, bvec, xW, LB * LT, LG4, LD);                      // 2048 blocks

  // ---- phase 2: recurrence over T ----
  k_init<<<(LB * LU + 255) / 256, 256, 0, stream>>>(h0, c0, masks, hm, c_ws, h_ws);
  for (int t = 0; t < LT; ++t) {
    k_rec_gemm<<<16, 256, 0, stream>>>(hm, Ut, xW, zbuf, t);    // 128 waves
    k_cell<<<(LB * LU + 255) / 256, 256, 0, stream>>>(zbuf, masks, c_ws, h_ws, hm, hs, t);
  }

  // ---- phase 3: logits = hs @ Wd + bd   (M=4096, N=32000, K=1024) ----
  k_gemm_bf16_lds<<<(LB * LT / BM) * (LV / BN), 256, 0, stream>>>(
      hs, Wdt, bd, out, LB * LT, LV, LU);                       // 16000 blocks

  // ---- tail: hT, cT ----
  k_finalize<<<(2 * LB * LU + 255) / 256, 256, 0, stream>>>(
      h_ws, c_ws, out + (size_t)LOGITS_ELEMS);
}
